// MultiHeadAttention_35141422416246
// MI455X (gfx1250) — matrix operand order repaired and checked by reading, not compiled, not run on hardware
//
#include <hip/hip_runtime.h>
#include <cstdint>
#include <cstddef>

#define DEV static __device__ __forceinline__

typedef __attribute__((ext_vector_type(16))) __bf16 bf16x16;
typedef __attribute__((ext_vector_type(8)))  float  f32x8;
typedef __attribute__((ext_vector_type(4)))  unsigned int v4u;
typedef __attribute__((ext_vector_type(8)))  int v8i_;
typedef __attribute__((ext_vector_type(4)))  int v4i_;

union FragB { unsigned int u[8]; bf16x16 v; };

static constexpr int Bc  = 4;
static constexpr int Sc  = 2048;
static constexpr int Dc  = 1024;   // D_IN == D_OUT
static constexpr int Hc  = 16;
static constexpr int HDc = 64;
static constexpr int Mc  = Bc * Sc;  // 8192 rows

#if __has_builtin(__builtin_amdgcn_s_wait_asynccnt)
#define WAIT_ASYNC(n) __builtin_amdgcn_s_wait_asynccnt(n)
#else
#define WAIT_ASYNC(n) asm volatile("s_wait_asynccnt %0" ::"i"(n) : "memory")
#endif

#if __has_builtin(__builtin_amdgcn_tensor_load_to_lds) && \
    __has_builtin(__builtin_amdgcn_s_wait_tensorcnt)
#define HAVE_TDM 1
#else
#define HAVE_TDM 0
#endif

// one per-lane 16B async copy: global -> LDS (gfx1250 async-DMA path, ASYNCcnt)
DEV void async_b128(unsigned lds_byte_addr, unsigned long long gaddr, int imm_off) {
  asm volatile("global_load_async_to_lds_b128 %0, %1, off offset:%2"
               :: "v"(lds_byte_addr), "v"(gaddr), "i"(imm_off)
               : "memory");
}

// ---- helpers -------------------------------------------------------------

DEV unsigned short f2bf(float f) {
  unsigned int u = __builtin_bit_cast(unsigned int, f);
  u += 0x7fffu + ((u >> 16) & 1u);       // round-to-nearest-even
  return (unsigned short)(u >> 16);
}
DEV unsigned int pack2(float a, float b) {
  return (unsigned int)f2bf(a) | ((unsigned int)f2bf(b) << 16);
}

DEV f32x8 wmma_bf16(const FragB& a, const FragB& b, f32x8 c) {
  return __builtin_amdgcn_wmma_f32_16x16x32_bf16(
      /*neg_a=*/false, a.v, /*neg_b=*/false, b.v,
      /*c_mod=*/(short)0, c, /*reuse_a=*/false, /*reuse_b=*/false);
}

// contiguous k order this code uses for A and B (not AMD's table order): lanes 0-15: K=0..15 ; lanes 16-31: K=16..31
DEV int kbA(int i, int hi) { return 2 * i + hi * 16; }

// ---- conversion kernels --------------------------------------------------

__global__ void cvt_bf16_pairs(const float* __restrict__ in,
                               unsigned int* __restrict__ out, int npair) {
  int i = blockIdx.x * blockDim.x + threadIdx.x;
  if (i < npair) out[i] = pack2(in[2 * i], in[2 * i + 1]);
}

// W [1024][1024] f32 (row = k, col = n) -> Wpk[512][1024] dwords: (k, k+1) pairs
__global__ void pack_w(const float* __restrict__ W, unsigned int* __restrict__ Wpk) {
  int idx = blockIdx.x * blockDim.x + threadIdx.x;   // kp*1024 + col
  int kp  = idx >> 10;
  int col = idx & 1023;
  float a = W[(2 * kp) * Dc + col];
  float b = W[(2 * kp + 1) * Dc + col];
  Wpk[idx] = pack2(a, b);
}

// ---- generic WMMA GEMM: C[8192x1024] = Xb @ W  ---------------------------
// B (weight) tile staged by the Tensor Data Mover (TENSORcnt), A (activation)
// tile staged by async global->LDS DMA (ASYNCcnt); both double-buffered.
// MODE 0: store bf16 headed row-major  [b][h][s][hd]          (Q, K)
// MODE 1: store bf16 headed pair-pack  [b][h][s/2][hd] dwords (V)
// MODE 2: store f32 + bias to dst      [row][col]             (final output)
template <int MODE>
__global__ void gemm_wmma(const unsigned short* __restrict__ X,
                          const unsigned int* __restrict__ Wpk,
                          void* __restrict__ dst,
                          const float* __restrict__ bias) {
  __shared__ unsigned int Atile[2][128 * 16];  // 2 x 8KB: 128 rows x 32 bf16
  __shared__ unsigned int Btile[2][16 * 64];   // 2 x 4KB: 32 k x 64 n (pairs)

  const int tid  = threadIdx.x;
  const int wave = tid >> 5;
  const int lane = tid & 31;
  const int hi   = lane >> 4;
  const int ln   = lane & 15;
  const int mbase = blockIdx.y * 128 + wave * 32;  // 4 waves x 32 rows
  const int nbase = blockIdx.x * 64;               // 64 cols per block

  // A tile: lane `tid` moves row (blockIdx.y*128 + tid), 64B = 4 x b128
  auto issue_atile = [&](int buf, int kk) {
    const unsigned short* g = X + (size_t)(blockIdx.y * 128 + tid) * Dc + kk;
    unsigned long long ga = (unsigned long long)(size_t)g;
    unsigned lds0 = (unsigned)(size_t)(void*)&Atile[buf][tid * 16];
    async_b128(lds0,      ga, 0);
    async_b128(lds0 + 16, ga, 16);
    async_b128(lds0 + 32, ga, 32);
    async_b128(lds0 + 48, ga, 48);
  };

#if HAVE_TDM
  // B tile via TDM: one 2D descriptor moves 16 rows x 64 dwords (4KB)
  auto issue_btile = [&](int buf, int kk) {
    unsigned long long ga =
        (unsigned long long)(size_t)(Wpk + (size_t)(kk >> 1) * Dc + nbase);
    unsigned lds = (unsigned)(size_t)(void*)&Btile[buf][0];
    v4u g0;
    g0.x = 1u;                                        // count=1, user D#
    g0.y = lds;                                       // lds_addr
    g0.z = (unsigned)ga;                              // global_addr[31:0]
    g0.w = (unsigned)((ga >> 32) & 0x1FFFFFFu)        // global_addr[56:32]
           | (2u << 30);                              // type = 2 (image)
    v8i_ g1;
    g1[0] = 0x00020000;   // data_size = 2 (4 bytes)
    g1[1] = 0x04000000;   // tensor_dim0[15:0] = 1024  (bits 63:48)
    g1[2] = 0x02000000;   // tensor_dim1[15:0] = 512   (bits 95:80)
    g1[3] = 0x00400000;   // tile_dim0 = 64            (bits 127:112)
    g1[4] = 0x00000010;   // tile_dim1 = 16            (bits 143:128)
    g1[5] = 0x00000400;   // tensor_dim0_stride = 1024 (bits 191:160)
    g1[6] = 0;
    g1[7] = 0;
    v4i_ z4;
    z4[0] = 0; z4[1] = 0; z4[2] = 0; z4[3] = 0;
    v8i_ z8;
    z8[0] = 0; z8[1] = 0; z8[2] = 0; z8[3] = 0;
    z8[4] = 0; z8[5] = 0; z8[6] = 0; z8[7] = 0;
    __builtin_amdgcn_tensor_load_to_lds(g0, g1, z4, z4, z8, 0);
  };
  constexpr int A_ISSUES = 4;
#else
  // fallback: per-lane async copies, 1024 dwords, 8 per thread
  auto issue_btile = [&](int buf, int kk) {
    const int idx0 = tid * 8;
    const int row  = idx0 >> 6;
    const int col  = idx0 & 63;
    const unsigned int* g = Wpk + (size_t)((kk >> 1) + row) * Dc + nbase + col;
    unsigned long long ga = (unsigned long long)(size_t)g;
    unsigned lds0 = (unsigned)(size_t)(void*)&Btile[buf][row * 64 + col];
    async_b128(lds0, ga, 0);
    async_b128(lds0 + 16, ga, 16);
  };
  constexpr int A_ISSUES = 6;   // 4 A + 2 B async issues per wave per tile
#endif

  f32x8 c[2][4];
#pragma unroll
  for (int mt = 0; mt < 2; ++mt)
#pragma unroll
    for (int t = 0; t < 4; ++t) c[mt][t] = f32x8{};

  // prologue: stage tile 0
#if HAVE_TDM
  if (wave == 0) issue_btile(0, 0);
#else
  issue_btile(0, 0);
#endif
  issue_atile(0, 0);
  int cur = 0;

  for (int kk = 0; kk < Dc; kk += 32) {
    const bool more = (kk + 32 < Dc);
    if (more) {
#if HAVE_TDM
      if (wave == 0) issue_btile(cur ^ 1, kk + 32);
#else
      issue_btile(cur ^ 1, kk + 32);
#endif
      issue_atile(cur ^ 1, kk + 32);
      WAIT_ASYNC(A_ISSUES);   // previous tile's async issues retired (in-order)
    } else {
      WAIT_ASYNC(0);
    }
#if HAVE_TDM
    if (wave == 0) {
      if (more) __builtin_amdgcn_s_wait_tensorcnt(1);
      else      __builtin_amdgcn_s_wait_tensorcnt(0);
    }
#endif
    __syncthreads();

    FragB a[2];
#pragma unroll
    for (int mt = 0; mt < 2; ++mt) {
      const int rowl = wave * 32 + mt * 16 + ln;
#pragma unroll
      for (int i = 0; i < 8; ++i)
        a[mt].u[i] = Atile[cur][rowl * 16 + (kbA(i, hi) >> 1)];
    }
#pragma unroll
    for (int t = 0; t < 4; ++t) {
      FragB bf;
#pragma unroll
      for (int i = 0; i < 8; ++i)
        bf.u[i] = Btile[cur][(i + hi * 8) * 64 + t * 16 + ln];
      c[0][t] = wmma_bf16(a[0], bf, c[0][t]);
      c[1][t] = wmma_bf16(a[1], bf, c[1][t]);
    }
    __syncthreads();   // all waves done reading tiles before they are rewritten
    cur ^= 1;
  }

  // store
#pragma unroll
  for (int mt = 0; mt < 2; ++mt) {
#pragma unroll
    for (int t = 0; t < 4; ++t) {
      const int col = nbase + t * 16 + ln;
      float bb = 0.0f;
      if (MODE == 2) bb = bias[col];
#pragma unroll
      for (int r = 0; r < 8; ++r) {
        const int row = mbase + mt * 16 + r + hi * 8;  // M = r (lanes<16) / r+8
        if (MODE == 2) {
          ((float*)dst)[(size_t)row * Dc + col] = c[mt][t][r] + bb;
        } else {
          const int b = row >> 11, s = row & (Sc - 1);
          const int h = col >> 6, hd = col & (HDc - 1);
          if (MODE == 0) {
            ((unsigned short*)dst)[(((size_t)(b * Hc + h)) * Sc + s) * HDc + hd] =
                f2bf(c[mt][t][r]);
          } else if (MODE == 1) {
            if ((r & 1) == 0) {
              const int sp = s >> 1;  // rows row,row+1 -> pair
              ((unsigned int*)dst)[(((size_t)(b * Hc + h)) * (Sc / 2) + sp) * HDc + hd] =
                  pack2(c[mt][t][r], c[mt][t][r + 1]);
            }
          }
        }
      }
    }
  }
}

// ---- flash attention: 1 wave per 16-row Q tile ---------------------------
__global__ void attn_wmma(const unsigned short* __restrict__ Qb,
                          const unsigned short* __restrict__ Kb,
                          const unsigned int* __restrict__ Vpk,
                          unsigned short* __restrict__ ctx) {
  const int nq    = Sc / 16;                  // 128 q-tiles per head
  const int bh    = blockIdx.x / nq;          // 0..63
  const int qbase = (blockIdx.x % nq) * 16;
  const int lane  = threadIdx.x;
  const int hi    = lane >> 4;
  const int ln    = lane & 15;

  const unsigned short* Qh = Qb + (size_t)bh * Sc * HDc;
  const unsigned short* Kh = Kb + (size_t)bh * Sc * HDc;
  const unsigned int*   Vh = Vpk + (size_t)bh * (Sc / 2) * HDc;

  __shared__ unsigned short Plds[16 * 32];

  // Q A-fragments for the two K=32 steps over hd=64
  FragB qa[2];
  {
    const unsigned int* rowp = (const unsigned int*)(Qh + (size_t)(qbase + ln) * HDc);
#pragma unroll
    for (int s = 0; s < 2; ++s)
#pragma unroll
      for (int i = 0; i < 8; ++i) qa[s].u[i] = rowp[(kbA(i, hi) + 32 * s) >> 1];
  }

  f32x8 acc[4];
#pragma unroll
  for (int t = 0; t < 4; ++t) acc[t] = f32x8{};
  float m[8], l[8];
#pragma unroll
  for (int r = 0; r < 8; ++r) { m[r] = -1.0e30f; l[r] = 0.0f; }

  const float SC = 0.18033688011112042f;  // (1/sqrt(64)) * log2(e)
  const int kend = qbase + 15;

  for (int kk = 0; kk <= kend; kk += 32) {
    // prefetch next K/V chunk into cache one iteration ahead
    if (kk + 32 <= kend) {
      int pr = kk + 32 + lane;
      if (pr > Sc - 1) pr = Sc - 1;
      __builtin_prefetch(Kh + (size_t)pr * HDc, 0, 1);
      __builtin_prefetch(Vh + (size_t)((kk + 32) / 2 + lane) * HDc, 0, 1);
    }

    // ---- scores: two 16x16 tiles (k cols kk..kk+15 and kk+16..kk+31)
    f32x8 sc[2];
#pragma unroll
    for (int t = 0; t < 2; ++t) {
      int krow = kk + t * 16 + ln;
      if (krow > Sc - 1) krow = Sc - 1;
      const unsigned int* kp = (const unsigned int*)(Kh + (size_t)krow * HDc);
      FragB kb0, kb1;
#pragma unroll
      for (int i = 0; i < 8; ++i) {
        kb0.u[i] = kp[i + hi * 8];         // hd = 2i + 16*hi
        kb1.u[i] = kp[i + hi * 8 + 16];    // hd = 2i + 16*hi + 32
      }
      f32x8 z = f32x8{};
      z = wmma_bf16(qa[0], kb0, z);
      z = wmma_bf16(qa[1], kb1, z);
      sc[t] = z;
    }

    // ---- scale + causal mask
    const bool needmask = (kk + 31 > qbase);
#pragma unroll
    for (int t = 0; t < 2; ++t) {
      const int kcol = kk + t * 16 + ln;
#pragma unroll
      for (int r = 0; r < 8; ++r) {
        float v = sc[t][r] * SC;
        if (needmask && (kcol > qbase + r + hi * 8)) v = -1.0e30f;
        sc[t][r] = v;
      }
    }

    // ---- online softmax per q row (rows live across 16-lane groups)
#pragma unroll
    for (int r = 0; r < 8; ++r) {
      float rm = fmaxf(sc[0][r], sc[1][r]);
#pragma unroll
      for (int msk = 1; msk < 16; msk <<= 1) rm = fmaxf(rm, __shfl_xor(rm, msk, 32));
      const float mn    = fmaxf(m[r], rm);
      const float alpha = exp2f(m[r] - mn);
      const float p0 = exp2f(sc[0][r] - mn);
      const float p1 = exp2f(sc[1][r] - mn);
      sc[0][r] = p0;
      sc[1][r] = p1;
      float rs = p0 + p1;
#pragma unroll
      for (int msk = 1; msk < 16; msk <<= 1) rs += __shfl_xor(rs, msk, 32);
      l[r] = l[r] * alpha + rs;
      m[r] = mn;
#pragma unroll
      for (int t = 0; t < 4; ++t) acc[t][r] *= alpha;
    }

    // ---- transpose P (C layout -> A layout) through LDS
#pragma unroll
    for (int t = 0; t < 2; ++t)
#pragma unroll
      for (int r = 0; r < 8; ++r)
        Plds[(r + hi * 8) * 32 + t * 16 + ln] = f2bf(sc[t][r]);
    __syncthreads();

    FragB pa;
    {
      const unsigned int* pp = (const unsigned int*)(Plds + ln * 32);
#pragma unroll
      for (int i = 0; i < 8; ++i) pa.u[i] = pp[kbA(i, hi) >> 1];
    }

    // ---- acc += P @ V  (4 hd tiles of 16 cols)
#pragma unroll
    for (int t = 0; t < 4; ++t) {
      FragB vb;
      const int n = t * 16 + ln;
#pragma unroll
      for (int i = 0; i < 8; ++i) {
        int sp = (kk >> 1) + i + hi * 8;
        if (sp > Sc / 2 - 1) sp = Sc / 2 - 1;
        vb.u[i] = Vh[(size_t)sp * HDc + n];
      }
      acc[t] = wmma_bf16(pa, vb, acc[t]);
    }
    __syncthreads();
  }

  // ---- normalize and store ctx as bf16 [b][s][h*64+hd]
  const int b = bh >> 4, h = bh & 15;
#pragma unroll
  for (int r = 0; r < 8; ++r) {
    const int s = qbase + r + hi * 8;
    const float inv = 1.0f / l[r];
#pragma unroll
    for (int t = 0; t < 4; ++t) {
      const int col = h * HDc + t * 16 + ln;
      ctx[((size_t)(b * Sc + s)) * Dc + col] = f2bf(acc[t][r] * inv);
    }
  }
}

// ---- host launcher -------------------------------------------------------

extern "C" void kernel_launch(void* const* d_in, const int* in_sizes, int n_in,
                              void* d_out, int out_size, void* d_ws, size_t ws_size,
                              hipStream_t stream) {
  (void)in_sizes; (void)n_in; (void)out_size; (void)ws_size;

  const float* x  = (const float*)d_in[0];
  const float* Wq = (const float*)d_in[1];
  const float* Wk = (const float*)d_in[2];
  const float* Wv = (const float*)d_in[3];
  const float* Wo = (const float*)d_in[4];
  const float* bo = (const float*)d_in[5];

  char* ws = (char*)d_ws;
  unsigned short* xb = (unsigned short*)ws;          ws += (size_t)Mc * Dc * 2;  // 16.78 MB
  unsigned int* wqpk = (unsigned int*)ws;            ws += (size_t)(Dc / 2) * Dc * 4;
  unsigned int* wkpk = (unsigned int*)ws;            ws += (size_t)(Dc / 2) * Dc * 4;
  unsigned int* wvpk = (unsigned int*)ws;            ws += (size_t)(Dc / 2) * Dc * 4;
  unsigned int* wopk = (unsigned int*)ws;            ws += (size_t)(Dc / 2) * Dc * 4;
  unsigned short* qb = (unsigned short*)ws;          ws += (size_t)Mc * HDc * Hc * 2;
  unsigned short* kb = (unsigned short*)ws;          ws += (size_t)Mc * HDc * Hc * 2;
  unsigned int* vpk  = (unsigned int*)ws;            ws += (size_t)Mc * HDc * Hc * 2;
  unsigned short* ctx = xb;  // reuse: x-bf16 dead after QKV projections

  // 1) x -> bf16
  {
    const int npair = Mc * Dc / 2;
    cvt_bf16_pairs<<<npair / 256, 256, 0, stream>>>(x, (unsigned int*)xb, npair);
  }
  // 2) pack weights into WMMA-B pair layout
  {
    const int n = (Dc / 2) * Dc;
    pack_w<<<n / 256, 256, 0, stream>>>(Wq, wqpk);
    pack_w<<<n / 256, 256, 0, stream>>>(Wk, wkpk);
    pack_w<<<n / 256, 256, 0, stream>>>(Wv, wvpk);
    pack_w<<<n / 256, 256, 0, stream>>>(Wo, wopk);
  }
  // 3) Q, K, V projections
  {
    dim3 grid(Dc / 64, Mc / 128), block(128);
    gemm_wmma<0><<<grid, block, 0, stream>>>(xb, wqpk, (void*)qb, nullptr);
    gemm_wmma<0><<<grid, block, 0, stream>>>(xb, wkpk, (void*)kb, nullptr);
    gemm_wmma<1><<<grid, block, 0, stream>>>(xb, wvpk, (void*)vpk, nullptr);
  }
  // 4) causal flash attention
  {
    const int blocks = Bc * Hc * (Sc / 16);  // 8192
    attn_wmma<<<blocks, 32, 0, stream>>>(qb, kb, vpk, ctx);
  }
  // 5) output projection + bias -> f32 d_out
  {
    dim3 grid(Dc / 64, Mc / 128), block(128);
    gemm_wmma<2><<<grid, block, 0, stream>>>(ctx, wopk, d_out, bo);
  }
}
